// WeightedInvariantMPNNLayer_62809601736821
// MI455X (gfx1250) — matrix operand order repaired and checked by reading, not compiled, not run on hardware
//
#include <hip/hip_runtime.h>
#include <hip/hip_bf16.h>

typedef __attribute__((ext_vector_type(16))) __bf16 v16bf;
typedef __attribute__((ext_vector_type(8)))  float  v8f;
typedef __attribute__((ext_vector_type(4)))  unsigned int v4u;
typedef __attribute__((ext_vector_type(4)))  float  v4f;
typedef __attribute__((ext_vector_type(4)))  int    v4i;

#define AS1 __attribute__((address_space(1)))
#define AS3 __attribute__((address_space(3)))

#define D_H   128
#define N_RBF 16

// LDS row strides (in halves), padded so the 16 B-fragment lanes hit all 64
// LDS banks conflict-free with ds_load_b128:
//   stride_dwords mod 64 must generate distinct multiples of 4 per lane.
#define KS_M1  296   // Wm1T rows: K used = 288 (272 padded), 592B = 148 dw, 148%64=20
#define KS_128 136   // 128-K matrices + SiLU tiles:          272B =  68 dw,  68%64=4
#define KS_H1  264   // Wh1T rows: K used = 256:              528B = 132 dw, 132%64=4

union Frag {
    v16bf bf;
    v4u   u[2];
    unsigned short s[16];
};

__device__ __forceinline__ unsigned short f2bf(float f) {
    union { __bf16 b; unsigned short u; } c;
    c.b = (__bf16)f;
    return c.u;
}

__device__ __forceinline__ v8f wmma_bf16(const Frag& a, const Frag& b, v8f c) {
    return __builtin_amdgcn_wmma_f32_16x16x32_bf16(false, a.bf, false, b.bf,
                                                   (short)0, c, false, false);
}

// Async global -> LDS 16B copy (CDNA5 ASYNCcnt path), with compile-safe fallback.
__device__ __forceinline__ void cp16_async(const void* g, void* l) {
#if __has_builtin(__builtin_amdgcn_global_load_async_to_lds_b128)
    __builtin_amdgcn_global_load_async_to_lds_b128(
        (AS1 v4i*)g, (AS3 v4i*)l, 0, 0);
#else
    *reinterpret_cast<v4u*>(l) = *reinterpret_cast<const v4u*>(g);
#endif
}

__device__ __forceinline__ void wait_async() {
#if __has_builtin(__builtin_amdgcn_s_wait_asynccnt)
    __builtin_amdgcn_s_wait_asynccnt(0);
#else
    asm volatile("s_wait_asynccnt 0" ::: "memory");
#endif
}

// ---------------- prep kernels ----------------

__global__ void k_cvt_bf16(const float* __restrict__ in, __bf16* __restrict__ out,
                           long long n) {
    long long i = (long long)blockIdx.x * blockDim.x + threadIdx.x;
    long long stride = (long long)gridDim.x * blockDim.x;
    for (; i < n; i += stride) out[i] = (__bf16)in[i];
}

// W[K][NC] row-major (fp32) -> WT[NC][KPAD] (bf16), zero-padded in K
__global__ void k_transpose_pad(const float* __restrict__ W, __bf16* __restrict__ WT,
                                int K, int KPAD, int NC) {
    int i = blockIdx.x * blockDim.x + threadIdx.x;
    if (i >= NC * KPAD) return;
    int col = i / KPAD, k = i % KPAD;
    WT[i] = (k < K) ? (__bf16)W[(long long)k * NC + col] : (__bf16)0.0f;
}

__global__ void k_zero(float* __restrict__ p, long long n) {
    long long i = (long long)blockIdx.x * blockDim.x + threadIdx.x;
    long long stride = (long long)gridDim.x * blockDim.x;
    for (; i < n; i += stride) p[i] = 0.0f;
}

// ---------------- edge message kernel ----------------
// 256 threads = 8 waves; each wave owns a 16-edge tile.
// Weights staged once per block into LDS via async copies.

#define EDGE_W1_BYTES (D_H * KS_M1 * 2)    // 75776
#define EDGE_W2_BYTES (D_H * KS_128 * 2)   // 34816
#define EDGE_S_OFF_H  (D_H * KS_M1 + D_H * KS_128)      // halves
#define EDGE_SMEM_B   ((EDGE_S_OFF_H + 8 * 16 * KS_128) * 2)

__launch_bounds__(256)
__global__ void edge_kernel(const float* __restrict__ x,
                            const long long* __restrict__ ei,
                            const float* __restrict__ ew,
                            const __bf16* __restrict__ h16,
                            const __bf16* __restrict__ Wm1T,
                            const float* __restrict__ bm1,
                            const __bf16* __restrict__ Wm2T,
                            const float* __restrict__ bm2,
                            float* __restrict__ agg,
                            int E)
{
    extern __shared__ unsigned short smem[];
    unsigned short* W1L = smem;                       // [128][KS_M1]
    unsigned short* W2L = smem + D_H * KS_M1;         // [128][KS_128]
    unsigned short* SL  = smem + EDGE_S_OFF_H;        // 8 waves x [16][KS_128]

    const int tid  = threadIdx.x;
    const int w    = tid >> 5;
    const int lane = tid & 31;
    const int half = lane >> 4;
    const int l15  = lane & 15;

    // ---- stage weights to LDS (async; overlapped with gather/RBF below) ----
    {
        const char* g1 = (const char*)Wm1T;
        const char* g2 = (const char*)Wm2T;
        char* l1 = (char*)W1L;
        char* l2 = (char*)W2L;
        for (int off = tid * 16; off < EDGE_W1_BYTES; off += 256 * 16)
            cp16_async(g1 + off, l1 + off);
        for (int off = tid * 16; off < EDGE_W2_BYTES; off += 256 * 16)
            cp16_async(g2 + off, l2 + off);
    }

    const long long e0 = (long long)blockIdx.x * 128 + (long long)w * 16;
    long long e = e0 + l15;
    if (e >= E) e = E - 1;                  // clamp: EXEC stays uniform for WMMA
    const long long s = ei[e];
    const long long d = ei[(long long)E + e];

    // per-edge Gaussian RBF (this lane-half needs rbf[half*8 .. half*8+7])
    float dx = x[s * 2 + 0] - x[d * 2 + 0];
    float dy = x[s * 2 + 1] - x[d * 2 + 1];
    float r  = sqrtf(dx * dx + dy * dy + 1e-8f);
    const float width = 1.4142135623730951f / 15.0f;
    const float gamma = 1.0f / (2.0f * width * width);
    float rv[8];
#pragma unroll
    for (int j = 0; j < 8; ++j) {
        float mu = (float)(half * 8 + j) * width;
        float t  = r - mu;
        rv[j] = __expf(-gamma * t * t);
    }

    // A fragments of m_in (16x288): K-tiles 0-3 = h[src], 4-7 = h[dst], 8 = rbf|pad
    Frag A[9];
    const __bf16* hs = h16 + s * D_H;
    const __bf16* hd = h16 + d * D_H;
#pragma unroll
    for (int kt = 0; kt < 4; ++kt) {
        int base = kt * 32 + half * 8;     // ISA A-layout: half splits K 0-7/16-23 vs 8-15/24-31
        A[kt    ].u[0] = *reinterpret_cast<const v4u*>(hs + base);
        A[kt    ].u[1] = *reinterpret_cast<const v4u*>(hs + base + 16);
        A[kt + 4].u[0] = *reinterpret_cast<const v4u*>(hd + base);
        A[kt + 4].u[1] = *reinterpret_cast<const v4u*>(hd + base + 16);
    }
#pragma unroll
    for (int i = 0; i < 8; ++i) { A[8].s[i] = f2bf(rv[i]); A[8].s[i + 8] = 0; }

    wait_async();
    __syncthreads();      // weights visible to all waves

    // GEMM1 (16x288 @ 288x128) + bias + SiLU -> LDS (bf16)
#pragma unroll
    for (int nt = 0; nt < 8; ++nt) {
        v8f c = {};
        const unsigned short* brow = W1L + (nt * 16 + l15) * KS_M1 + half * 8;
#pragma unroll
        for (int kt = 0; kt < 9; ++kt) {
            Frag B;   // lane holds K 8h..8h+7 and 16+8h..16+8h+7 of column nt*16+l15
            B.u[0] = *reinterpret_cast<const v4u*>(brow + kt * 32);
            B.u[1] = *reinterpret_cast<const v4u*>(brow + kt * 32 + 16);
            c = wmma_bf16(A[kt], B, c);
        }
        int cg = nt * 16 + l15;
        float bb = bm1[cg];
#pragma unroll
        for (int vr = 0; vr < 8; ++vr) {
            int M = vr + half * 8;          // C-layout row
            float v  = c[vr] + bb;
            float sv = v * (1.0f / (1.0f + __expf(-v)));   // SiLU
            SL[w * 16 * KS_128 + M * KS_128 + cg] = f2bf(sv);
        }
    }
    __syncthreads();

    // A2 fragments from LDS (C-layout -> A-layout reshape)
    Frag A2[4];
#pragma unroll
    for (int kt = 0; kt < 4; ++kt) {
        const unsigned short* sp = SL + w * 16 * KS_128 + l15 * KS_128 + kt * 32 + half * 8;
        A2[kt].u[0] = *reinterpret_cast<const v4u*>(sp);
        A2[kt].u[1] = *reinterpret_cast<const v4u*>(sp + 16);
    }

    // GEMM2 (16x128 @ 128x128) + bias, scale by edge_weight, scatter-add
#pragma unroll
    for (int nt = 0; nt < 8; ++nt) {
        v8f c2 = {};
        const unsigned short* brow = W2L + (nt * 16 + l15) * KS_128 + half * 8;
#pragma unroll
        for (int kt = 0; kt < 4; ++kt) {
            Frag B;
            B.u[0] = *reinterpret_cast<const v4u*>(brow + kt * 32);
            B.u[1] = *reinterpret_cast<const v4u*>(brow + kt * 32 + 16);
            c2 = wmma_bf16(A2[kt], B, c2);
        }
        int cg = nt * 16 + l15;
        float bb = bm2[cg];
#pragma unroll
        for (int vr = 0; vr < 8; ++vr) {
            int M = vr + half * 8;
            long long em = e0 + M;
            if (em < E) {
                float mv = (c2[vr] + bb) * ew[em];
                int dn = (int)ei[(long long)E + em];
                atomicAdd(agg + (long long)dn * D_H + cg, mv);   // L2-resident atomics
            }
        }
    }
}

// ---------------- node update kernel ----------------

#define NODE_W1_BYTES (D_H * KS_H1 * 2)    // 67584
#define NODE_W2_BYTES (D_H * KS_128 * 2)   // 34816
#define NODE_S_OFF_H  (D_H * KS_H1 + D_H * KS_128)
#define NODE_SMEM_B   ((NODE_S_OFF_H + 8 * 16 * KS_128) * 2)

__launch_bounds__(256)
__global__ void node_kernel(const float* __restrict__ h,
                            const __bf16* __restrict__ h16,
                            const float* __restrict__ agg,
                            const __bf16* __restrict__ Wh1T,
                            const float* __restrict__ bh1,
                            const __bf16* __restrict__ Wh2T,
                            const float* __restrict__ bh2,
                            const float* __restrict__ lng,
                            const float* __restrict__ lnb,
                            float* __restrict__ out,
                            int N)
{
    extern __shared__ unsigned short smem[];
    unsigned short* W1L = smem;                       // [128][KS_H1]
    unsigned short* W2L = smem + D_H * KS_H1;         // [128][KS_128]
    unsigned short* SL  = smem + NODE_S_OFF_H;

    const int tid  = threadIdx.x;
    const int w    = tid >> 5;
    const int lane = tid & 31;
    const int half = lane >> 4;
    const int l15  = lane & 15;

    {
        const char* g1 = (const char*)Wh1T;
        const char* g2 = (const char*)Wh2T;
        char* l1 = (char*)W1L;
        char* l2 = (char*)W2L;
        for (int off = tid * 16; off < NODE_W1_BYTES; off += 256 * 16)
            cp16_async(g1 + off, l1 + off);
        for (int off = tid * 16; off < NODE_W2_BYTES; off += 256 * 16)
            cp16_async(g2 + off, l2 + off);
    }

    const long long n0 = (long long)blockIdx.x * 128 + (long long)w * 16;
    long long n = n0 + l15;
    if (n >= N) n = N - 1;

    // A fragments of u_in = [h | agg] (16x256)
    Frag A[8];
    const __bf16* hp = h16 + n * D_H;
    const float*  ap = agg + n * D_H;
#pragma unroll
    for (int kt = 0; kt < 4; ++kt) {
        int base = kt * 32 + half * 8;
        A[kt].u[0] = *reinterpret_cast<const v4u*>(hp + base);
        A[kt].u[1] = *reinterpret_cast<const v4u*>(hp + base + 16);
        v4f f0 = *reinterpret_cast<const v4f*>(ap + base);
        v4f f1 = *reinterpret_cast<const v4f*>(ap + base + 4);
        v4f f2 = *reinterpret_cast<const v4f*>(ap + base + 16);
        v4f f3 = *reinterpret_cast<const v4f*>(ap + base + 20);
#pragma unroll
        for (int i = 0; i < 4; ++i) {
            A[4 + kt].s[i     ] = f2bf(f0[i]);
            A[4 + kt].s[4 + i ] = f2bf(f1[i]);
            A[4 + kt].s[8 + i ] = f2bf(f2[i]);
            A[4 + kt].s[12 + i] = f2bf(f3[i]);
        }
    }

    wait_async();
    __syncthreads();

    // GEMM1 + SiLU -> LDS
#pragma unroll
    for (int nt = 0; nt < 8; ++nt) {
        v8f c = {};
        const unsigned short* brow = W1L + (nt * 16 + l15) * KS_H1 + half * 8;
#pragma unroll
        for (int kt = 0; kt < 8; ++kt) {
            Frag B;
            B.u[0] = *reinterpret_cast<const v4u*>(brow + kt * 32);
            B.u[1] = *reinterpret_cast<const v4u*>(brow + kt * 32 + 16);
            c = wmma_bf16(A[kt], B, c);
        }
        int cg = nt * 16 + l15;
        float bb = bh1[cg];
#pragma unroll
        for (int vr = 0; vr < 8; ++vr) {
            int M = vr + half * 8;
            float v  = c[vr] + bb;
            float sv = v * (1.0f / (1.0f + __expf(-v)));
            SL[w * 16 * KS_128 + M * KS_128 + cg] = f2bf(sv);
        }
    }
    __syncthreads();

    Frag A2[4];
#pragma unroll
    for (int kt = 0; kt < 4; ++kt) {
        const unsigned short* sp = SL + w * 16 * KS_128 + l15 * KS_128 + kt * 32 + half * 8;
        A2[kt].u[0] = *reinterpret_cast<const v4u*>(sp);
        A2[kt].u[1] = *reinterpret_cast<const v4u*>(sp + 16);
    }

    // GEMM2 + bias + residual; keep z in registers for LayerNorm
    float z[8][8];   // [nt][vr]
#pragma unroll
    for (int nt = 0; nt < 8; ++nt) {
        v8f c2 = {};
        const unsigned short* brow = W2L + (nt * 16 + l15) * KS_128 + half * 8;
#pragma unroll
        for (int kt = 0; kt < 4; ++kt) {
            Frag B;
            B.u[0] = *reinterpret_cast<const v4u*>(brow + kt * 32);
            B.u[1] = *reinterpret_cast<const v4u*>(brow + kt * 32 + 16);
            c2 = wmma_bf16(A2[kt], B, c2);
        }
        int cg = nt * 16 + l15;
        float bb = bh2[cg];
#pragma unroll
        for (int vr = 0; vr < 8; ++vr) {
            long long nn = n0 + vr + half * 8;
            if (nn >= N) nn = N - 1;
            z[nt][vr] = h[nn * D_H + cg] + c2[vr] + bb;
        }
    }

    // LayerNorm: row M's 128 values live in 16 lanes (same half) x 8 nt tiles.
    float mean[8], inv[8];
#pragma unroll
    for (int vr = 0; vr < 8; ++vr) {
        float s1 = 0.0f, s2 = 0.0f;
#pragma unroll
        for (int nt = 0; nt < 8; ++nt) { float zz = z[nt][vr]; s1 += zz; s2 += zz * zz; }
        for (int m = 1; m < 16; m <<= 1) {   // xor stays within 16-lane half (wave32)
            s1 += __shfl_xor(s1, m, 32);
            s2 += __shfl_xor(s2, m, 32);
        }
        float mu  = s1 * (1.0f / 128.0f);
        float var = s2 * (1.0f / 128.0f) - mu * mu;
        mean[vr] = mu;
        inv[vr]  = rsqrtf(var + 1e-5f);
    }

#pragma unroll
    for (int nt = 0; nt < 8; ++nt) {
        int cg = nt * 16 + l15;
        float gg = lng[cg], bb = lnb[cg];
#pragma unroll
        for (int vr = 0; vr < 8; ++vr) {
            long long nn = n0 + vr + half * 8;
            if (nn < N)
                out[nn * D_H + cg] = (z[nt][vr] - mean[vr]) * inv[vr] * gg + bb;
        }
    }
}

// ---------------- launch ----------------

extern "C" void kernel_launch(void* const* d_in, const int* in_sizes, int n_in,
                              void* d_out, int out_size, void* d_ws, size_t ws_size,
                              hipStream_t stream) {
    const float*     h   = (const float*)d_in[0];
    const float*     x   = (const float*)d_in[1];
    const long long* ei  = (const long long*)d_in[2];  // int64 edge_index [2,E]
    const float*     ew  = (const float*)d_in[3];
    const float*     Wm1 = (const float*)d_in[4];
    const float*     bm1 = (const float*)d_in[5];
    const float*     Wm2 = (const float*)d_in[6];
    const float*     bm2 = (const float*)d_in[7];
    const float*     Wh1 = (const float*)d_in[8];
    const float*     bh1 = (const float*)d_in[9];
    const float*     Wh2 = (const float*)d_in[10];
    const float*     bh2 = (const float*)d_in[11];
    const float*     lng = (const float*)d_in[12];
    const float*     lnb = (const float*)d_in[13];

    const int N = in_sizes[0] / D_H;
    const int E = in_sizes[3];

    char* ws = (char*)d_ws;
    auto alloc = [&](size_t bytes) {
        char* p = ws;
        ws += (bytes + 255) & ~(size_t)255;
        return p;
    };
    __bf16* h16  = (__bf16*)alloc((size_t)N * D_H * 2);
    __bf16* Wm1T = (__bf16*)alloc((size_t)D_H * KS_M1 * 2);
    __bf16* Wm2T = (__bf16*)alloc((size_t)D_H * KS_128 * 2);
    __bf16* Wh1T = (__bf16*)alloc((size_t)D_H * KS_H1 * 2);
    __bf16* Wh2T = (__bf16*)alloc((size_t)D_H * KS_128 * 2);
    float*  agg  = (float*)alloc((size_t)N * D_H * 4);

    // Allow >64KB dynamic LDS (WGP has 320KB). Not a stream op; capture-safe.
    (void)hipFuncSetAttribute(reinterpret_cast<const void*>(edge_kernel),
                              hipFuncAttributeMaxDynamicSharedMemorySize, EDGE_SMEM_B);
    (void)hipFuncSetAttribute(reinterpret_cast<const void*>(node_kernel),
                              hipFuncAttributeMaxDynamicSharedMemorySize, NODE_SMEM_B);

    const long long hn = (long long)N * D_H;
    k_cvt_bf16<<<512, 256, 0, stream>>>(h, h16, hn);
    k_transpose_pad<<<(D_H * KS_M1  + 255) / 256, 256, 0, stream>>>(Wm1, Wm1T, 2 * D_H + N_RBF, KS_M1,  D_H);
    k_transpose_pad<<<(D_H * KS_128 + 255) / 256, 256, 0, stream>>>(Wm2, Wm2T, D_H,             KS_128, D_H);
    k_transpose_pad<<<(D_H * KS_H1  + 255) / 256, 256, 0, stream>>>(Wh1, Wh1T, 2 * D_H,         KS_H1,  D_H);
    k_transpose_pad<<<(D_H * KS_128 + 255) / 256, 256, 0, stream>>>(Wh2, Wh2T, D_H,             KS_128, D_H);
    k_zero<<<512, 256, 0, stream>>>(agg, hn);

    edge_kernel<<<(E + 127) / 128, 256, EDGE_SMEM_B, stream>>>(x, ei, ew, h16, Wm1T, bm1,
                                                               Wm2T, bm2, agg, E);
    node_kernel<<<(N + 127) / 128, 256, NODE_SMEM_B, stream>>>(h, h16, agg, Wh1T, bh1,
                                                               Wh2T, bh2, lng, lnb,
                                                               (float*)d_out, N);
}
